// GraphAttentionLayer_80668075753786
// MI455X (gfx1250) — compile-verified, matrix operand order checked by reading, not run on hardware
//
#include <hip/hip_runtime.h>
#include <hip/hip_bf16.h>

#define B_ 16
#define N_ 512
#define D_ 64

typedef __attribute__((ext_vector_type(16))) _Float16 v16h;
typedef __attribute__((ext_vector_type(8)))  _Float16 v8h;
typedef __attribute__((ext_vector_type(8)))  float    v8f;
typedef __attribute__((ext_vector_type(4)))  float    v4f;

__device__ __forceinline__ v8h ld8h(const _Float16* p) { return *(const v8h*)p; }

__device__ __forceinline__ v16h cat16(v8h lo, v8h hi) {
    v16h r;
#pragma unroll
    for (int e = 0; e < 8; ++e) { r[e] = lo[e]; r[e + 8] = hi[e]; }
    return r;
}

// load 8 consecutive f32 (32B aligned), convert to 8 x f16
__device__ __forceinline__ v8h cvt8(const float* p) {
    v4f a = *(const v4f*)p;
    v4f b = *(const v4f*)(p + 4);
    v8h r;
#pragma unroll
    for (int e = 0; e < 4; ++e) { r[e] = (_Float16)a[e]; r[e + 4] = (_Float16)b[e]; }
    return r;
}

__device__ __forceinline__ v8f wmma_f16(v16h a, v16h b, v8f c) {
    return __builtin_amdgcn_wmma_f32_16x16x32_f16(false, a, false, b, (short)0, c,
                                                  false, false);
}

// Native CDNA5 tanh (TRANS32 op).
__device__ __forceinline__ float fast_tanh(float x) {
#if __has_builtin(__builtin_amdgcn_tanhf)
    return __builtin_amdgcn_tanhf(x);
#elif __has_builtin(__builtin_amdgcn_tanh_f32)
    return __builtin_amdgcn_tanh_f32(x);
#else
    float r;
    asm("v_tanh_f32 %0, %1\n\tv_nop" : "=v"(r) : "v"(x));
    return r;
#endif
}

// Branchless exp via native v_exp_f32 (2^x): args are <= 0 here (softmax / SELU).
__device__ __forceinline__ float fast_exp(float x) {
    return __builtin_amdgcn_exp2f(x * 1.4426950408889634f);
}

// ---------------- kernel 0: f32 -> f16 copies (row-major + transposed) ----------
__global__ void gat_prep(const float* __restrict__ x,
                         _Float16* __restrict__ xh, _Float16* __restrict__ xhT) {
    int row = blockIdx.x;               // b*N + j
    int b = row >> 9, j = row & (N_ - 1);
    int d = threadIdx.x;
    float v = x[(size_t)row * D_ + d];
    xh[(size_t)row * D_ + d] = (_Float16)v;
    xhT[((size_t)b * D_ + d) * N_ + j] = (_Float16)v;
}

// ---------------- kernel 1: attention + projections + BN partials ----------------
__global__ __launch_bounds__(512) void gat_main(
    const _Float16* __restrict__ xh, const _Float16* __restrict__ xhT,
    const float* __restrict__ Wap, const float* __restrict__ bap,
    const float* __restrict__ attw,
    const float* __restrict__ Wwith, const float* __restrict__ bwith,
    const float* __restrict__ Wwo, const float* __restrict__ bwo,
    float* __restrict__ xout, float* __restrict__ bn_sum, float* __restrict__ bn_sq)
{
    __shared__ float    s_lds[16][N_];      // scores, one row per wave
    __shared__ _Float16 e_lds[16][N_];      // exp(s - max), softmax numerators
    __shared__ _Float16 agg_lds[16][D_];    // attention-weighted aggregate
    __shared__ float    denom_lds[16];

    const int blk  = blockIdx.x;
    const int b    = blk >> 5;
    const int bi0  = (blk & 31) * 16;       // i-tile base
    const int tid  = threadIdx.x;
    const int w    = tid >> 5;              // wave id (0..15), owns i = bi0 + w
    const int lane = tid & 31;
    const int ocol = lane & 15;             // column within 16-wide tile
    const int hb   = (lane >> 4) * 8;       // K / row offset for high half-wave

    // ===== phase 1: scores s[i, j] for i = bi0 + w, all j ===================
    const _Float16* xrow_i = xh + (size_t)(b * N_ + bi0 + w) * D_;

    // A-operand C_i(o, d) = x_i[d] * W_att_proj[o][d], resident in registers.
    // Lane = row o (within tile), K packing: d = ks*32 + (e>>3)*16 + hb + (e&7).
    v16h Cb[4][2];
#pragma unroll
    for (int ks = 0; ks < 2; ++ks) {
        int d0 = ks * 32 + hb;
        v8h xa = ld8h(xrow_i + d0);
        v8h xb = ld8h(xrow_i + d0 + 16);
#pragma unroll
        for (int t = 0; t < 4; ++t) {
            const float* wr = Wap + (t * 16 + ocol) * D_;
            v8h wa = cvt8(wr + d0);
            v8h wb = cvt8(wr + d0 + 16);
#pragma unroll
            for (int e = 0; e < 8; ++e) {
                Cb[t][ks][e]     = wa[e] * xa[e];
                Cb[t][ks][e + 8] = wb[e] * xb[e];
            }
        }
    }
    // D'[o, j] layout: VGPR r -> o = t*16 + r + hb, lane -> j. Bias folds into
    // the WMMA C operand; att weights become a per-lane register vector.
    v8f biasC[4], wtv[4];
#pragma unroll
    for (int t = 0; t < 4; ++t) {
        const float* bp = bap + t * 16 + hb;
        const float* wp = attw + t * 16 + hb;
#pragma unroll
        for (int r = 0; r < 8; ++r) { biasC[t][r] = bp[r]; wtv[t][r] = wp[r]; }
    }

    // B-operand fragments: 16 columns x_j (same packing as A rows).
    const _Float16* xbase = xh + (size_t)(b * N_ + ocol) * D_;   // + J*16*D_
    const int d0lo = hb;                     // ks = 0 chunk base
    const int d0hi = 32 + hb;                // ks = 1 chunk base

#define LOAD_FRAG(X, p)                                                   \
    do {                                                                  \
        (X)[0] = cat16(ld8h((p) + d0lo), ld8h((p) + d0lo + 16));          \
        (X)[1] = cat16(ld8h((p) + d0hi), ld8h((p) + d0hi + 16));          \
    } while (0)

#define SCORE_TILE(X, sdst)                                               \
    do {                                                                  \
        float s_ = 0.f;                                                   \
        _Pragma("unroll")                                                 \
        for (int t = 0; t < 4; ++t) {                                     \
            v8f acc = wmma_f16(Cb[t][0], (X)[0], biasC[t]);               \
            acc = wmma_f16(Cb[t][1], (X)[1], acc);                        \
            _Pragma("unroll")                                             \
            for (int r = 0; r < 8; ++r)                                   \
                s_ += fast_tanh(acc[r]) * wtv[t][r];                      \
        }                                                                 \
        s_ += __shfl_xor(s_, 16, 32);                                     \
        if (lane < 16) (sdst) = s_;                                       \
    } while (0)

    v16h X0[2], X1[2];
    LOAD_FRAG(X0, xbase);
    LOAD_FRAG(X1, xbase + 16 * D_);
    for (int J = 0; J < 32; J += 2) {
        // tile J from X0, then refill X0 with tile J+2 (clamped on last iter)
        SCORE_TILE(X0, s_lds[w][J * 16 + lane]);
        LOAD_FRAG(X0, xbase + (size_t)((J + 2 < 32) ? J + 2 : 30) * (16 * D_));
        // tile J+1 from X1, then refill X1 with tile J+3 (clamped on last iter)
        SCORE_TILE(X1, s_lds[w][(J + 1) * 16 + lane]);
        LOAD_FRAG(X1, xbase + (size_t)((J + 3 < 32) ? J + 3 : 31) * (16 * D_));
    }
#undef SCORE_TILE
#undef LOAD_FRAG
    __syncthreads();

    // ===== phase 2: softmax over j (per wave, own row) =======================
    {
        float vals[16];
        float m = -1e30f;
#pragma unroll
        for (int k = 0; k < 16; ++k) {
            vals[k] = s_lds[w][lane + 32 * k];
            m = fmaxf(m, vals[k]);
        }
#pragma unroll
        for (int msk = 16; msk >= 1; msk >>= 1) m = fmaxf(m, __shfl_xor(m, msk, 32));
        float dsum = 0.f;
#pragma unroll
        for (int k = 0; k < 16; ++k) {
            float e = fast_exp(vals[k] - m);
            dsum += e;
            e_lds[w][lane + 32 * k] = (_Float16)e;
        }
#pragma unroll
        for (int msk = 16; msk >= 1; msk >>= 1) dsum += __shfl_xor(dsum, msk, 32);
        if (lane == 0) denom_lds[w] = dsum;
    }
    __syncthreads();

    // ===== phase 3: agg(16 x 64) = E(16 x 512) @ X(512 x 64), waves 0..3 =====
    if (w < 4) {
        const int t = w;                    // d-tile
        v8f acc = {};
        const _Float16* xcol = xhT + ((size_t)(b * D_) + t * 16 + ocol) * N_;
        const _Float16* erow = &e_lds[ocol][0];
        for (int ks = 0; ks < 16; ++ks) {
            int j0 = ks * 32 + hb;
            v16h Af = cat16(ld8h(erow + j0), ld8h(erow + j0 + 16));
            v16h Bf = cat16(ld8h(xcol + j0), ld8h(xcol + j0 + 16));
            acc = wmma_f16(Af, Bf, acc);
        }
#pragma unroll
        for (int r = 0; r < 8; ++r) {
            int row = r + hb;
            float inv = 1.f / denom_lds[row];
            agg_lds[row][t * 16 + ocol] = (_Float16)(acc[r] * inv);
        }
    }
    __syncthreads();

    // ===== phase 4: out = agg@Wwith^T + x@Wwo^T + biases; BN partials ========
    if (w < 4) {
        const int t = w;                    // o-tile
        const int o = t * 16 + ocol;
        v8f acc = {};
        const _Float16* arow = &agg_lds[ocol][0];
        const _Float16* xrow = xh + (size_t)(b * N_ + bi0 + ocol) * D_;
        const float* w1 = Wwith + o * D_;
        const float* w2 = Wwo + o * D_;
#pragma unroll
        for (int ks = 0; ks < 2; ++ks) {
            int d0 = ks * 32 + hb;
            v16h A1 = cat16(ld8h(arow + d0), ld8h(arow + d0 + 16));
            v16h B1 = cat16(cvt8(w1 + d0), cvt8(w1 + d0 + 16));
            acc = wmma_f16(A1, B1, acc);
            v16h A2 = cat16(ld8h(xrow + d0), ld8h(xrow + d0 + 16));
            v16h B2 = cat16(cvt8(w2 + d0), cvt8(w2 + d0 + 16));
            acc = wmma_f16(A2, B2, acc);
        }
        float bias = bwith[o] + bwo[o];
        float csum = 0.f, csq = 0.f;
#pragma unroll
        for (int r = 0; r < 8; ++r) {
            float v = acc[r] + bias;
            int row = bi0 + r + hb;
            xout[((size_t)(b * N_) + row) * D_ + o] = v;
            csum += v;
            csq  += v * v;
        }
        csum += __shfl_xor(csum, 16, 32);   // combine the two 8-row halves
        csq  += __shfl_xor(csq, 16, 32);
        if (hb == 0) {
            atomicAdd(&bn_sum[o], csum);
            atomicAdd(&bn_sq[o], csq);
        }
    }
}

// ---------------- kernel 2: BN scale/shift ---------------------------------
__global__ void gat_bnfin(const float* __restrict__ bn_sum, const float* __restrict__ bn_sq,
                          const float* __restrict__ gamma, const float* __restrict__ beta,
                          float* __restrict__ scale, float* __restrict__ shift) {
    int o = threadIdx.x;
    const float invn = 1.f / (float)(B_ * N_);
    float mean = bn_sum[o] * invn;
    float var  = bn_sq[o] * invn - mean * mean;
    float sc   = gamma[o] * rsqrtf(var + 1e-5f);
    scale[o] = sc;
    shift[o] = beta[o] - mean * sc;
}

// ---------------- kernel 3: apply BN + SELU --------------------------------
__global__ void gat_bnselu(const float* __restrict__ xout,
                           const float* __restrict__ scale, const float* __restrict__ shift,
                           float* __restrict__ out) {
    int idx = blockIdx.x * 256 + threadIdx.x;
    int o = idx & (D_ - 1);
    float y = xout[idx] * scale[o] + shift[o];
    const float lam = 1.0507009873554805f, alpha = 1.6732632423543772f;
    out[idx] = y > 0.f ? lam * y
                       : lam * alpha * (__builtin_amdgcn_exp2f(y * 1.4426950408889634f) - 1.f);
}

extern "C" void kernel_launch(void* const* d_in, const int* in_sizes, int n_in,
                              void* d_out, int out_size, void* d_ws, size_t ws_size,
                              hipStream_t stream) {
    const float* x     = (const float*)d_in[0];
    const float* Wap   = (const float*)d_in[1];
    const float* bap   = (const float*)d_in[2];
    const float* attw  = (const float*)d_in[3];
    const float* Wwith = (const float*)d_in[4];
    const float* bwith = (const float*)d_in[5];
    const float* Wwo   = (const float*)d_in[6];
    const float* bwo   = (const float*)d_in[7];
    const float* gamma = (const float*)d_in[8];
    const float* beta  = (const float*)d_in[9];
    float* out = (float*)d_out;

    char* ws = (char*)d_ws;
    _Float16* xh   = (_Float16*)(ws);                      // 1 MB
    _Float16* xhT  = (_Float16*)(ws + (1u << 20));         // 1 MB
    float*    xout = (float*)(ws + (2u << 20));            // 2 MB
    float*    bn_sum = (float*)(ws + (4u << 20));          // 256 B
    float*    bn_sq  = (float*)(ws + (4u << 20) + 256);    // 256 B
    float*    scale  = (float*)(ws + (4u << 20) + 512);
    float*    shift  = (float*)(ws + (4u << 20) + 768);

    hipMemsetAsync((void*)bn_sum, 0, 512, stream);

    gat_prep<<<B_ * N_, D_, 0, stream>>>(x, xh, xhT);
    gat_main<<<B_ * (N_ / 16), 512, 0, stream>>>(xh, xhT, Wap, bap, attw,
                                                 Wwith, bwith, Wwo, bwo,
                                                 xout, bn_sum, bn_sq);
    gat_bnfin<<<1, D_, 0, stream>>>(bn_sum, bn_sq, gamma, beta, scale, shift);
    gat_bnselu<<<(B_ * N_ * D_) / 256, 256, 0, stream>>>(xout, scale, shift, out);
}
